// HopkinsLoss_45810121179403
// MI455X (gfx1250) — compile-verified, matrix operand order checked by reading, not run on hardware
//
#include <hip/hip_runtime.h>

typedef __attribute__((ext_vector_type(16))) __bf16 v16bf;
typedef __attribute__((ext_vector_type(8)))  float  v8f;

union BF16x16 { unsigned u[8]; uint4 q[2]; v16bf v; };

#define FBIG 3.402823466e38f

__device__ __forceinline__ unsigned bf_rne(float f) {
  unsigned u = __float_as_uint(f);
  return (u + 0x7FFFu + ((u >> 16) & 1u)) >> 16;   // round-to-nearest-even bf16 bits
}

// ---------------- one-time f32 -> split bf16 (hi/lo) conversion ----------------
__global__ void hop_cvt_bf16(const float* __restrict__ src, unsigned short* __restrict__ hi,
                             unsigned short* __restrict__ lo, size_t n4) {
  size_t i = (size_t)blockIdx.x * blockDim.x + threadIdx.x;
  if (i >= n4) return;
  const float4 f = ((const float4*)src)[i];
  ushort4 H, L;
  unsigned b;
  b = bf_rne(f.x); H.x = (unsigned short)b; L.x = (unsigned short)bf_rne(f.x - __uint_as_float(b << 16));
  b = bf_rne(f.y); H.y = (unsigned short)b; L.y = (unsigned short)bf_rne(f.y - __uint_as_float(b << 16));
  b = bf_rne(f.z); H.z = (unsigned short)b; L.z = (unsigned short)bf_rne(f.z - __uint_as_float(b << 16));
  b = bf_rne(f.w); H.w = (unsigned short)b; L.w = (unsigned short)bf_rne(f.w - __uint_as_float(b << 16));
  ((ushort4*)hi)[i] = H;
  ((ushort4*)lo)[i] = L;
}

// ---------------- column min/max of X (two pass) ----------------
__global__ void hop_colminmax_partial(const float* __restrict__ X, float* __restrict__ pmin,
                                      float* __restrict__ pmax, int N, int D, int rowsPerBlock) {
  int c = blockIdx.x * blockDim.x + threadIdx.x;
  if (c >= D) return;
  int r0 = blockIdx.y * rowsPerBlock;
  int r1 = r0 + rowsPerBlock; if (r1 > N) r1 = N;
  float mn = FBIG, mx = -FBIG;
  for (int r = r0; r < r1; ++r) {
    float v = X[(size_t)r * D + c];
    mn = fminf(mn, v); mx = fmaxf(mx, v);
  }
  pmin[(size_t)blockIdx.y * D + c] = mn;
  pmax[(size_t)blockIdx.y * D + c] = mx;
}

__global__ void hop_colminmax_final(const float* __restrict__ pmin, const float* __restrict__ pmax,
                                    float* __restrict__ xmin, float* __restrict__ xmax, int D, int RB) {
  int c = blockIdx.x * blockDim.x + threadIdx.x;
  if (c >= D) return;
  float mn = FBIG, mx = -FBIG;
  for (int rb = 0; rb < RB; ++rb) {
    mn = fminf(mn, pmin[(size_t)rb * D + c]);
    mx = fmaxf(mx, pmax[(size_t)rb * D + c]);
  }
  xmin[c] = mn; xmax[c] = mx;
}

// ---------------- row squared norms of X ----------------
__global__ void hop_rownorm(const float* __restrict__ X, float* __restrict__ xnorm, int D) {
  __shared__ float red[256];
  int row = blockIdx.x;
  const float* p = X + (size_t)row * D;
  float s = 0.f;
  for (int k = threadIdx.x; k < D; k += blockDim.x) { float v = p[k]; s += v * v; }
  red[threadIdx.x] = s; __syncthreads();
  for (int o = 128; o > 0; o >>= 1) {
    if (threadIdx.x < o) red[threadIdx.x] += red[threadIdx.x + o];
    __syncthreads();
  }
  if (threadIdx.x == 0) xnorm[row] = red[0];
}

// ------- build A = [Y ; X_tilde] (padded) directly as split bf16, plus row norms -------
__global__ void hop_buildA(const float* __restrict__ X, const float* __restrict__ U,
                           const int* __restrict__ idx, const float* __restrict__ xmin,
                           const float* __restrict__ xmax,
                           unsigned short* __restrict__ Ahi, unsigned short* __restrict__ Alo,
                           float* __restrict__ anorm, int D, int M, int Mp) {
  __shared__ float red[256];
  int row = blockIdx.x;
  unsigned short* ah = Ahi + (size_t)row * D;
  unsigned short* al = Alo + (size_t)row * D;
  float s = 0.f;
  if (row < Mp) {
    int i = row;
    if (i < M) {
      const float* u = U + (size_t)i * D;
      for (int k = threadIdx.x; k < D; k += blockDim.x) {
        float y = (xmin[k] - xmax[k]) * u[k] + xmax[k];
        unsigned hb = bf_rne(y);
        ah[k] = (unsigned short)hb;
        al[k] = (unsigned short)bf_rne(y - __uint_as_float(hb << 16));
        s += y * y;
      }
    } else {
      for (int k = threadIdx.x; k < D; k += blockDim.x) { ah[k] = 0; al[k] = 0; }
    }
  } else {
    int ti = row - Mp;
    if (ti < M) {
      const float* x = X + (size_t)idx[ti] * D;
      for (int k = threadIdx.x; k < D; k += blockDim.x) {
        float v = x[k];
        unsigned hb = bf_rne(v);
        ah[k] = (unsigned short)hb;
        al[k] = (unsigned short)bf_rne(v - __uint_as_float(hb << 16));
        s += v * v;
      }
    } else {
      for (int k = threadIdx.x; k < D; k += blockDim.x) { ah[k] = 0; al[k] = 0; }
    }
  }
  red[threadIdx.x] = s; __syncthreads();
  for (int o = 128; o > 0; o >>= 1) {
    if (threadIdx.x < o) red[threadIdx.x] += red[threadIdx.x + o];
    __syncthreads();
  }
  if (threadIdx.x == 0) anorm[row] = red[0];
}

__global__ void hop_initmin(unsigned* __restrict__ minv, int n) {
  int i = blockIdx.x * blockDim.x + threadIdx.x;
  if (i < n) minv[i] = 0x7F800000u;  // +inf
}

// ---------------- fused split-bf16 WMMA distance GEMM + row-min ----------------
// Block: 32 A-rows (2 WMMA tiles) x 128 X-cols (8 waves x 16 cols).
// All operands pre-converted to bf16 hi/lo: fragments are 2x b128 loads each.
// No LDS, no barriers; A lines shared by the 8 waves through WGP$/L2.
__global__ __launch_bounds__(256) void hop_dist_min(
    const unsigned short* __restrict__ Ahi, const unsigned short* __restrict__ Alo,
    const unsigned short* __restrict__ Xhi, const unsigned short* __restrict__ Xlo,
    const float* __restrict__ anorm, const float* __restrict__ xnorm,
    const int* __restrict__ idx, unsigned* __restrict__ minv,
    int N, int D, int M, int Mp) {
  const int tid  = threadIdx.x;
  const int lane = tid & 31;
  const int wave = tid >> 5;
  const int h    = lane >> 4;   // K-half select (16-bit WMMA layout)
  const int lm   = lane & 15;   // A-row / B-col within tile

  const int rowBase = blockIdx.x * 32;
  const int colBase = blockIdx.y * 128;
  const int j = colBase + wave * 16 + lm;      // this lane's B column == X row
  const bool jvalid = (j < N);
  const size_t jc = jvalid ? (size_t)j : 0;    // clamped: loads always legal, result masked later

  // fragment base pointers (rows are 16B-aligned: D % 8 == 0, 2B elements)
  const uint4* pBh  = (const uint4*)(Xhi + jc * D);
  const uint4* pBl  = (const uint4*)(Xlo + jc * D);
  const uint4* pA0h = (const uint4*)(Ahi + (size_t)(rowBase + lm) * D);
  const uint4* pA0l = (const uint4*)(Alo + (size_t)(rowBase + lm) * D);
  const uint4* pA1h = (const uint4*)(Ahi + (size_t)(rowBase + 16 + lm) * D);
  const uint4* pA1l = (const uint4*)(Alo + (size_t)(rowBase + 16 + lm) * D);

  v8f acc0 = {};
  v8f acc1 = {};

  const int nk = D >> 5;    // K steps of 32
#pragma unroll 2
  for (int kk = 0; kk < nk; ++kk) {
    // 16-bit WMMA operand layout: within a 32-K window (16 dwords = 4 uint4),
    // half h uses uint4 {h, h+2} (K 0..7/16..23 for h=0, K 8..15/24..31 for h=1).
    const int q0 = kk * 4 + h;
    const int q1 = q0 + 2;
    BF16x16 bh, bl, a0h, a0l, a1h, a1l;
    bh.q[0]  = pBh[q0];  bh.q[1]  = pBh[q1];
    bl.q[0]  = pBl[q0];  bl.q[1]  = pBl[q1];
    a0h.q[0] = pA0h[q0]; a0h.q[1] = pA0h[q1];
    a0l.q[0] = pA0l[q0]; a0l.q[1] = pA0l[q1];
    a1h.q[0] = pA1h[q0]; a1h.q[1] = pA1h[q1];
    a1l.q[0] = pA1l[q0]; a1l.q[1] = pA1l[q1];
    // split-bf16: hi*hi + hi*lo + lo*hi accumulated in f32 (~fp32 accuracy)
    acc0 = __builtin_amdgcn_wmma_f32_16x16x32_bf16(false, a0h.v, false, bh.v, (short)0, acc0, false, false);
    acc0 = __builtin_amdgcn_wmma_f32_16x16x32_bf16(false, a0h.v, false, bl.v, (short)0, acc0, false, false);
    acc0 = __builtin_amdgcn_wmma_f32_16x16x32_bf16(false, a0l.v, false, bh.v, (short)0, acc0, false, false);
    acc1 = __builtin_amdgcn_wmma_f32_16x16x32_bf16(false, a1h.v, false, bh.v, (short)0, acc1, false, false);
    acc1 = __builtin_amdgcn_wmma_f32_16x16x32_bf16(false, a1h.v, false, bl.v, (short)0, acc1, false, false);
    acc1 = __builtin_amdgcn_wmma_f32_16x16x32_bf16(false, a1l.v, false, bh.v, (short)0, acc1, false, false);
  }

  // epilogue: d2 = |a|^2 + |x|^2 - 2*dot, self-exclusion, 16-lane min, atomicMin
  const float xn = jvalid ? xnorm[jc] : 0.f;
#pragma unroll
  for (int t = 0; t < 2; ++t) {
    int rbase = rowBase + t * 16 + h * 8;
#pragma unroll
    for (int r = 0; r < 8; ++r) {
      int row = rbase + r;
      float dot = (t == 0) ? acc0[r] : acc1[r];
      float d2 = fmaxf(anorm[row] + xn - 2.f * dot, 0.f);
      if (!jvalid) d2 = FBIG;
      if (row >= Mp) {                 // X_tilde half: exclude self (2-NN)
        int ti = row - Mp;
        if (ti < M && idx[ti] == j) d2 = FBIG;
      }
      d2 = fminf(d2, __shfl_xor(d2, 1, 32));
      d2 = fminf(d2, __shfl_xor(d2, 2, 32));
      d2 = fminf(d2, __shfl_xor(d2, 4, 32));
      d2 = fminf(d2, __shfl_xor(d2, 8, 32));
      if (lm == 0) atomicMin(&minv[row], __float_as_uint(d2));  // non-neg float == uint order
    }
  }
}

// ---------------- final scalar: H and |target - H| ----------------
__global__ void hop_finalize(const unsigned* __restrict__ minv, float* __restrict__ out,
                             int M, int Mp) {
  __shared__ float red[256];
  float su = 0.f, sw = 0.f;
  for (int i = threadIdx.x; i < M; i += blockDim.x) {
    su += sqrtf(__uint_as_float(minv[i]));
    sw += sqrtf(__uint_as_float(minv[Mp + i]));
  }
  red[threadIdx.x] = su; __syncthreads();
  for (int o = 128; o > 0; o >>= 1) { if (threadIdx.x < o) red[threadIdx.x] += red[threadIdx.x + o]; __syncthreads(); }
  float SU = red[0]; __syncthreads();
  red[threadIdx.x] = sw; __syncthreads();
  for (int o = 128; o > 0; o >>= 1) { if (threadIdx.x < o) red[threadIdx.x] += red[threadIdx.x + o]; __syncthreads(); }
  if (threadIdx.x == 0) {
    float SW = red[0];
    float H = SU / (SU + SW + 1e-12f);
    out[0] = fabsf(0.01f - H);
  }
}

extern "C" void kernel_launch(void* const* d_in, const int* in_sizes, int n_in,
                              void* d_out, int out_size, void* d_ws, size_t ws_size,
                              hipStream_t stream) {
  const float* X   = (const float*)d_in[0];
  const float* U   = (const float*)d_in[1];
  const int*   idx = (const int*)d_in[2];

  const int M  = in_sizes[2];
  const int D  = in_sizes[1] / M;
  const int N  = in_sizes[0] / D;
  const int Mp = ((M + 15) / 16) * 16;
  const int rowsA = 2 * Mp;

  // workspace layout (byte offsets, 256B aligned segments)
  char* w = (char*)d_ws;
  size_t off = 0;
  auto alloc = [&](size_t bytes) -> void* {
    void* p = w + off;
    off += bytes;
    off = (off + 255) & ~(size_t)255;
    return p;
  };
  unsigned short* Xhi = (unsigned short*)alloc((size_t)N * D * 2);
  unsigned short* Xlo = (unsigned short*)alloc((size_t)N * D * 2);
  unsigned short* Ahi = (unsigned short*)alloc((size_t)rowsA * D * 2);
  unsigned short* Alo = (unsigned short*)alloc((size_t)rowsA * D * 2);
  float* xnorm = (float*)alloc((size_t)N * 4);
  float* anorm = (float*)alloc((size_t)rowsA * 4);
  float* xmin  = (float*)alloc((size_t)D * 4);
  float* xmax  = (float*)alloc((size_t)D * 4);
  const int RB = 64;
  float* pmin  = (float*)alloc((size_t)RB * D * 4);
  float* pmax  = (float*)alloc((size_t)RB * D * 4);
  unsigned* minv = (unsigned*)alloc((size_t)rowsA * 4);

  // 1) pre-convert X to split bf16
  size_t n4 = (size_t)N * D / 4;
  hop_cvt_bf16<<<(unsigned)((n4 + 255) / 256), 256, 0, stream>>>(X, Xhi, Xlo, n4);

  // 2) bounding box + row norms
  const int rowsPerBlock = (N + RB - 1) / RB;
  dim3 gmm((D + 255) / 256, RB);
  hop_colminmax_partial<<<gmm, 256, 0, stream>>>(X, pmin, pmax, N, D, rowsPerBlock);
  hop_colminmax_final<<<(D + 255) / 256, 256, 0, stream>>>(pmin, pmax, xmin, xmax, D, RB);
  hop_rownorm<<<N, 256, 0, stream>>>(X, xnorm, D);

  // 3) build A = [Y ; X_tilde] in split bf16 + norms
  hop_buildA<<<rowsA, 256, 0, stream>>>(X, U, idx, xmin, xmax, Ahi, Alo, anorm, D, M, Mp);
  hop_initmin<<<(rowsA + 255) / 256, 256, 0, stream>>>(minv, rowsA);

  // 4) fused WMMA distance + min
  dim3 gd(rowsA / 32, (N + 127) / 128);
  hop_dist_min<<<gd, 256, 0, stream>>>(Ahi, Alo, Xhi, Xlo, anorm, xnorm, idx, minv, N, D, M, Mp);

  // 5) scalar finish
  hop_finalize<<<1, 256, 0, stream>>>(minv, (float*)d_out, M, Mp);
}
